// TemporalAdjLearner_71347996721374
// MI455X (gfx1250) — compile-verified, matrix operand order checked by reading, not run on hardware
//
#include <hip/hip_runtime.h>
#include <hip/hip_bf16.h>

typedef __attribute__((ext_vector_type(16))) __bf16 v16bf;
typedef __attribute__((ext_vector_type(8)))  float  v8f;
typedef __attribute__((ext_vector_type(4)))  float  v4f;
typedef __attribute__((ext_vector_type(2)))  unsigned int v2u;

union ABf { v16bf v; unsigned int d[8]; unsigned short u[16]; };

__device__ __forceinline__ unsigned short f2bf(float f) {
  unsigned u = __float_as_uint(f);
  unsigned r = u + 0x7FFFu + ((u >> 16) & 1u);   // round-to-nearest-even
  return (unsigned short)(r >> 16);
}

// N=8192, T=16, IN=1024, KD=128, TOPK=8
#define NROWS 8192
#define INDIM 1024
#define KDIM  128
#define SCALE 0.088388347648318447f   // 1/sqrt(128)

// --------------------------------------------------------------------------
// Kernel 0: convert Wq/Wk (f32 [1024][128]) to bf16, pre-swizzled into the
// WMMA B-operand per-lane layout so GEMM loads are contiguous 32B per lane.
// offset = ((((nt*32 + kc)*32 + lane)*8 + v)*2 + e), nt: 16-col tile (0..7),
// kc: 32-deep K chunk (0..31).  kk = (lane>=16)*8 + (v>=4)*16 + (v&3)*2 + e.
// --------------------------------------------------------------------------
__global__ void __launch_bounds__(256) k_swizzle_w(
    const float* __restrict__ Wq, const float* __restrict__ Wk,
    unsigned short* __restrict__ WqB, unsigned short* __restrict__ WkB) {
  int gid = blockIdx.x * 256 + threadIdx.x;   // 0 .. 262143 (two matrices)
  int sel = gid >> 17;
  int r   = gid & 131071;
  int e  = r & 1;
  int v  = (r >> 1) & 7;
  int l  = (r >> 4) & 31;
  int kc = (r >> 9) & 31;
  int nt = (r >> 14) & 7;
  int n  = nt * 16 + (l & 15);
  int kk = ((l >> 4) & 1) * 8 + ((v >> 2) & 1) * 16 + (v & 3) * 2 + e;
  int k  = kc * 32 + kk;
  const float* W = sel ? Wk : Wq;
  unsigned short* O = sel ? WkB : WqB;
  O[r] = f2bf(W[k * KDIM + n]);
}

// --------------------------------------------------------------------------
// Kernel 1: fused temporal mean-pool + Q/K projection (bf16 WMMA, f32 accum).
// Block handles 16 rows. Pool uses 128-bit non-temporal loads (U is a 512MB
// one-shot stream; keep it out of L2 so Ksw stays resident for kernel 2).
// Pooled tile staged in LDS as packed bf16 pairs. Q stored swizzled as future
// WMMA A-operand; K stored swizzled as future B-operand (mirrored layouts).
// --------------------------------------------------------------------------
__global__ void __launch_bounds__(256) k_pool_proj(
    const float* __restrict__ U,
    const unsigned short* __restrict__ WqB, const unsigned short* __restrict__ WkB,
    const float* __restrict__ bq, const float* __restrict__ bk,
    unsigned short* __restrict__ Qsw, unsigned short* __restrict__ Ksw) {
  __shared__ unsigned int pool[16 * INDIM / 2];   // 16 rows x 1024 bf16 (32KB)
  int tid = threadIdx.x;
  int b   = blockIdx.x;
  int r0  = b * 16;

  // ---- mean pool: 16 rows x 1024 dims, b128 NT loads, bf16-packed to LDS ----
  const v4f* U4 = (const v4f*)U;
  #pragma unroll 2
  for (int i = 0; i < 16; i++) {
    int p   = tid + i * 256;          // quad index 0..4095
    int row = p >> 8;                 // 256 quads per pooled row
    int dq  = p & 255;
    int base = (r0 + row) * (16 * INDIM / 4) + dq;  // v4f index, +256 per t
    v4f s = {0.f, 0.f, 0.f, 0.f};
    #pragma unroll
    for (int t = 0; t < 16; t++) {
      v4f q = __builtin_nontemporal_load(&U4[base + t * (INDIM / 4)]);
      s += q;
    }
    s *= 0.0625f;
    v2u packed;
    packed.x = (unsigned)f2bf(s.x) | ((unsigned)f2bf(s.y) << 16);
    packed.y = (unsigned)f2bf(s.z) | ((unsigned)f2bf(s.w) << 16);
    *(v2u*)&pool[(row * INDIM + dq * 4) >> 1] = packed;
  }
  __syncthreads();

  int lane = tid & 31, w = tid >> 5;
  int n0 = w * 16;
  int rowA = lane & 15;
  int baseKK = ((lane >> 4) & 1) * 8;

  for (int mat = 0; mat < 2; mat++) {
    const unsigned short* WB = mat ? WkB : WqB;
    const float* bias        = mat ? bk  : bq;
    unsigned short* Osw      = mat ? Ksw : Qsw;

    v8f c = {};
    for (int kc = 0; kc < 32; kc++) {
      ABf a, bm;
      #pragma unroll
      for (int v = 0; v < 8; v++) {
        int k0 = kc * 32 + baseKK + ((v >> 2) & 1) * 16 + (v & 3) * 2;
        a.d[v] = pool[(rowA * INDIM + k0) >> 1];
      }
      bm = *(const ABf*)(WB + (((w * 32 + kc) * 32 + lane) * 16));
      c = __builtin_amdgcn_wmma_f32_16x16x32_bf16(
            false, a.v, false, bm.v, (short)0, c, false, false);
    }

    // add bias, scatter-store into swizzled operand layout
    int col = n0 + (lane & 15);     // feature index d in 0..127
    float bv = bias[col];
    int kc2 = col >> 5, kk2 = col & 31;
    int half = (kk2 >> 3) & 1;
    int vv   = ((kk2 & 7) >> 1) + ((kk2 >> 4) & 1) * 4;
    int e2   = kk2 & 1;
    #pragma unroll
    for (int r = 0; r < 8; r++) {
      int rowT  = r + ((lane >> 4) << 3);
      int lane2 = half * 16 + rowT;
      int off = (((b * 4 + kc2) * 32 + lane2) * 8 + vv) * 2 + e2;
      Osw[off] = f2bf(c[r] + bv);
    }
  }
}

// --------------------------------------------------------------------------
// Kernel 2: fused scores GEMM (bf16 WMMA) + online softmax + running top-8.
// Block = 16 query rows; 8 waves sweep all 512 key tiles (K matrix = 2MB bf16,
// L2-resident). Never materializes the 8192x8192 score matrix.
// --------------------------------------------------------------------------
#define NWAVES 8
__global__ void __launch_bounds__(256) k_scores_topk(
    const unsigned short* __restrict__ Qsw, const unsigned short* __restrict__ Ksw,
    float* __restrict__ out_vals, int* __restrict__ out_rows,
    int* __restrict__ out_cols) {
  __shared__ float tv[NWAVES][16][8];
  __shared__ int   ti[NWAVES][16][8];
  __shared__ float rm[NWAVES][16];
  __shared__ float rl[NWAVES][16];

  int tid = threadIdx.x, lane = tid & 31, w = tid >> 5;
  int mt = blockIdx.x;

  for (int i = tid; i < NWAVES * 16 * 8; i += 256) {
    ((float*)tv)[i] = -__builtin_inff();
    ((int*)ti)[i] = 0;
  }
  __syncthreads();

  // A operands: this block's 16 Q rows, all 128 features (4 chunks of 32)
  ABf a[4];
  #pragma unroll
  for (int kc = 0; kc < 4; kc++)
    a[kc] = *(const ABf*)(Qsw + (((mt * 4 + kc) * 32 + lane) * 16));

  float m[8], lsum[8];
  #pragma unroll
  for (int r = 0; r < 8; r++) { m[r] = -__builtin_inff(); lsum[r] = 0.f; }
  int rhalf = (lane >> 4) << 3;

  for (int it = 0; it < 64; it++) {
    int nt = it * 8 + w;                      // 16-column key tile
    v8f c = {};
    #pragma unroll
    for (int kc = 0; kc < 4; kc++) {
      ABf bm = *(const ABf*)(Ksw + (((nt * 4 + kc) * 32 + lane) * 16));
      c = __builtin_amdgcn_wmma_f32_16x16x32_bf16(
            false, a[kc].v, false, bm.v, (short)0, c, false, false);
    }
    #pragma unroll
    for (int r = 0; r < 8; r++) {
      float v = c[r] * SCALE;
      int row = r + rhalf;
      // online softmax accumulation (per lane-slot partial)
      float mo = m[r];
      if (v > mo) { lsum[r] = lsum[r] * __expf(mo - v) + 1.0f; m[r] = v; }
      else        { lsum[r] += __expf(v - mo); }
      // running per-wave top-8 (ballot + serialized single-lane insert)
      float thr = tv[w][row][7];
      unsigned mask = (unsigned)__ballot(v > thr);
      while (mask) {
        int src = __ffs(mask) - 1; mask &= mask - 1;
        float cv = __shfl(v, src);
        if (lane == src) {
          int crow = r + ((src >> 4) << 3);
          int ccol = nt * 16 + (src & 15);
          float* Lv = tv[w][crow]; int* Li = ti[w][crow];
          if (cv > Lv[7]) {
            float pv = cv; int pi = ccol;
            #pragma unroll
            for (int k = 0; k < 8; k++) {
              float q = Lv[k]; int qi = Li[k];
              bool gt = pv > q;
              if (gt) { Lv[k] = pv; Li[k] = pi; pv = q; pi = qi; }
            }
          }
        }
      }
    }
  }

  // reduce (m, l) across each 16-lane half (each half owns 8 rows)
  #pragma unroll
  for (int r = 0; r < 8; r++) {
    float mm = m[r], ll = lsum[r];
    #pragma unroll
    for (int off = 1; off < 16; off <<= 1) {
      float m2 = __shfl_xor(mm, off, 16);
      float l2 = __shfl_xor(ll, off, 16);
      float mn = fmaxf(mm, m2);
      ll = ll * __expf(mm - mn) + l2 * __expf(m2 - mn);
      mm = mn;
    }
    if ((lane & 15) == 0) { rm[w][r + rhalf] = mm; rl[w][r + rhalf] = ll; }
  }
  __syncthreads();

  // final merge: one thread per row merges 8 wave partials + 64 candidates
  if (tid < 16) {
    int row = tid;
    float M = -__builtin_inff(), L = 0.f;
    for (int ww = 0; ww < NWAVES; ww++) {
      float m2 = rm[ww][row], l2 = rl[ww][row];
      float mn = fmaxf(M, m2);
      L = L * __expf(M - mn) + l2 * __expf(m2 - mn);
      M = mn;
    }
    float invL = 1.0f / L;
    int grow = mt * 16 + row;
    for (int k = 0; k < 8; k++) {
      float best = -__builtin_inff(); int bw = 0, bs = 0;
      for (int ww = 0; ww < NWAVES; ww++)
        for (int s = 0; s < 8; s++) {
          float x = tv[ww][row][s];
          if (x > best) { best = x; bw = ww; bs = s; }
        }
      tv[bw][row][bs] = -__builtin_inff();
      out_vals[grow * 8 + k] = __expf(best - M) * invL;
      out_rows[grow * 8 + k] = grow;
      out_cols[grow * 8 + k] = ti[bw][row][bs];
    }
  }
}

extern "C" void kernel_launch(void* const* d_in, const int* in_sizes, int n_in,
                              void* d_out, int out_size, void* d_ws, size_t ws_size,
                              hipStream_t stream) {
  const float* U  = (const float*)d_in[0];
  const float* Wq = (const float*)d_in[1];
  const float* bq = (const float*)d_in[2];
  const float* Wk = (const float*)d_in[3];
  const float* bk = (const float*)d_in[4];

  unsigned short* WqB = (unsigned short*)d_ws;            // 1024*128 bf16
  unsigned short* WkB = WqB + INDIM * KDIM;               // 1024*128 bf16
  unsigned short* Qsw = WkB + INDIM * KDIM;               // 8192*128 bf16
  unsigned short* Ksw = Qsw + NROWS * KDIM;               // 8192*128 bf16

  float* out_vals = (float*)d_out;                        // [65536] f32
  int*   outi     = (int*)d_out;
  int*   out_rows = outi + NROWS * 8;                     // [65536] i32
  int*   out_cols = outi + 2 * NROWS * 8;                 // [65536] i32

  k_swizzle_w  <<<1024, 256, 0, stream>>>(Wq, Wk, WqB, WkB);
  k_pool_proj  <<<NROWS / 16, 256, 0, stream>>>(U, WqB, WkB, bq, bk, Qsw, Ksw);
  k_scores_topk<<<NROWS / 16, 256, 0, stream>>>(Qsw, Ksw, out_vals, out_rows, out_cols);
}
